// Downsample_40664750358849
// MI455X (gfx1250) — matrix operand order repaired and checked by reading, not compiled, not run on hardware
//
#include <hip/hip_runtime.h>
#include <hip/hip_bf16.h>

typedef __attribute__((ext_vector_type(16))) __bf16        v16bf;
typedef __attribute__((ext_vector_type(8)))  float         v8f;
typedef __attribute__((ext_vector_type(4)))  unsigned int  v4u;

union Frag {
    v4u   u[2];
    v16bf v;
};

__device__ __forceinline__ unsigned short f2bf(float f) {
    unsigned int u = __builtin_bit_cast(unsigned int, f);
    unsigned int r = u + 0x7FFFu + ((u >> 16) & 1u);   // round-to-nearest-even
    return (unsigned short)(r >> 16);
}

// ---------------------------------------------------------------------------
// Pre-pass: repack conv weights (48,96,3,3) fp32 -> bf16 [khkw][co][ci]
// so A-fragments are contiguous-in-K per (khkw, co).
// ---------------------------------------------------------------------------
__global__ void repack_weights(const float* __restrict__ W,
                               unsigned short* __restrict__ wb, int total) {
    int i = blockIdx.x * blockDim.x + threadIdx.x;
    if (i >= total) return;
    int khkw = i % 9;
    int ci   = (i / 9) % 96;
    int co   = i / (9 * 96);
    wb[(khkw * 48 + co) * 96 + ci] = f2bf(W[i]);
}

// ---------------------------------------------------------------------------
// Fused conv3x3 (implicit GEMM via v_wmma_f32_16x16x32_bf16) + PixelUnshuffle.
// One block: (b, h, 128-wide row segment). 4 waves; each wave owns TWO
// 16-pixel N-subtiles x all 3 M-tiles -> 6 WMMAs per k-step off 2 B-frags.
// ---------------------------------------------------------------------------
__global__ __launch_bounds__(128)
void Downsample_conv_unshuffle(const float* __restrict__ x,
                               const unsigned short* __restrict__ wb,
                               float* __restrict__ out) {
    constexpr int CI = 96, H = 256, Wd = 256, NT = 128, NC = NT + 2;

    __shared__ unsigned short xs[3 * NC * CI];   // 74880 B, bf16 tile [row][col][ci]

    const int blk  = blockIdx.x;
    const int wseg = blk & 1;            // 2 segments of 128 pixels
    const int h    = (blk >> 1) & 255;
    const int b    = blk >> 9;
    const int w0   = wseg * NT;

    const int tid = threadIdx.x;

    // ---- stage x slab into LDS as bf16 (zero-padded borders) ----
    const float* xb = x + (size_t)b * CI * H * Wd;
    const int total = 3 * CI * NC;
    for (int idx = tid; idx < total; idx += 128) {
        int c  = idx % NC;
        int ci = (idx / NC) % CI;
        int r  = idx / (NC * CI);
        int hin = h - 1 + r;
        int win = w0 - 1 + c;
        float v = 0.0f;
        if ((unsigned)hin < (unsigned)H && (unsigned)win < (unsigned)Wd)
            v = xb[(ci * H + hin) * Wd + win];
        xs[(r * NC + c) * CI + ci] = f2bf(v);
    }
    __syncthreads();

    const int lane = tid & 31;           // wave32
    const int wave = tid >> 5;           // 0..3
    const int p0   = wave * 32;          // two N-subtiles: p0, p0+16
    const int lmod = lane & 15;
    const int lhi  = lane >> 4;

    v8f acc[3][2] = {{v8f{}, v8f{}}, {v8f{}, v8f{}}, {v8f{}, v8f{}}};

    #pragma unroll
    for (int khkw = 0; khkw < 9; ++khkw) {
        const int kh = khkw / 3;         // compile-time after unroll
        const int kw = khkw % 3;
        #pragma unroll
        for (int kc = 0; kc < 96; kc += 32) {
            // ---- two B fragments (K=32 x N=16): lane%16 = n, lane/16 = K-half
            Frag fb0, fb1;
            {
                int colp  = p0 + lmod + kw;                        // padded col
                int bbase = (kh * NC + colp) * CI + kc + lhi * 16; // halves
                const v4u* bp = reinterpret_cast<const v4u*>(&xs[bbase]);
                fb0.u[0] = bp[0];
                fb0.u[1] = bp[1];
                const v4u* bq = reinterpret_cast<const v4u*>(&xs[bbase + 16 * CI]);
                fb1.u[0] = bq[0];
                fb1.u[1] = bq[1];
            }
            // ---- 3 M-tiles of A (weights) from L2-resident repack; each A
            //      fragment feeds two WMMAs (both N-subtiles) ----
            #pragma unroll
            for (int mt = 0; mt < 3; ++mt) {
                Frag fa;
                int abase = (khkw * 48 + mt * 16 + lmod) * 96 + kc + lhi * 16;
                const v4u* ap = reinterpret_cast<const v4u*>(&wb[abase]);
                fa.u[0] = ap[0];     // K = kc + g*16  .. +8
                fa.u[1] = ap[1];     // K = kc + g*16 + 8 .. +8  (+16 bytes)
                acc[mt][0] = __builtin_amdgcn_wmma_f32_16x16x32_bf16(
                    false, fa.v, false, fb0.v, (short)0, acc[mt][0], false, false);
                acc[mt][1] = __builtin_amdgcn_wmma_f32_16x16x32_bf16(
                    false, fa.v, false, fb1.v, (short)0, acc[mt][1], false, false);
            }
        }
    }

    // ---- store with fused PixelUnshuffle(2):
    // out[b, co*4 + (h&1)*2 + (w&1), h/2, w/2] = y[b, co, h, w]
    #pragma unroll
    for (int ns = 0; ns < 2; ++ns) {
        const int wpix = w0 + p0 + ns * 16 + lmod;
        const int ohw  = (h >> 1) * 128 + (wpix >> 1);
        const int ocl  = (h & 1) * 2 + (wpix & 1);
        #pragma unroll
        for (int mt = 0; mt < 3; ++mt) {
            #pragma unroll
            for (int r = 0; r < 8; ++r) {
                int co = mt * 16 + r + lhi * 8;
                int oc = co * 4 + ocl;
                out[((size_t)(b * 192 + oc) << 14) + ohw] = acc[mt][ns][r];
            }
        }
    }
}

// ---------------------------------------------------------------------------
extern "C" void kernel_launch(void* const* d_in, const int* in_sizes, int n_in,
                              void* d_out, int out_size, void* d_ws, size_t ws_size,
                              hipStream_t stream) {
    const float* x = (const float*)d_in[0];          // (8,96,256,256) fp32
    const float* W = (const float*)d_in[1];          // (48,96,3,3)   fp32
    float* out = (float*)d_out;                      // (8,192,128,128) fp32
    unsigned short* wb = (unsigned short*)d_ws;      // 41472 bf16 = 82944 B

    const int wtotal = 48 * 96 * 9;
    repack_weights<<<(wtotal + 255) / 256, 256, 0, stream>>>(W, wb, wtotal);

    // blocks: b(8) * h(256) * wseg(2) = 4096 ; 128 threads = 4 wave32
    Downsample_conv_unshuffle<<<4096, 128, 0, stream>>>(x, wb, out);
}
